// MultiHeadAttention_54889682043438
// MI455X (gfx1250) — matrix operand order repaired and checked by reading, compiled, not run on hardware
//
#include <hip/hip_runtime.h>

// ---------------------------------------------------------------------------
// MI455X (gfx1250) fused multi-head attention forward.
// f32 inputs converted to bf16 once; all matmuls stream bf16 operands
// straight from memory into v_wmma_f32_16x16x32_bf16 (f32 accumulate).
// V is pre-transposed so every WMMA operand is a pure b128 vector load.
// ---------------------------------------------------------------------------

typedef __attribute__((ext_vector_type(16))) __bf16 v16bf;
typedef __attribute__((ext_vector_type(8)))  __bf16 v8bf;
typedef __attribute__((ext_vector_type(4)))  __bf16 v4bf;
typedef __attribute__((ext_vector_type(8)))  float  v8f;

static constexpr int Bsz  = 4;
static constexpr int Sseq = 2048;
static constexpr int Din  = 1024;
static constexpr int Dhid = 1024;
static constexpr int NH   = 16;
static constexpr int HDm  = 64;       // head dim
static constexpr int LDQ  = 3 * Dhid; // qkv row stride (3072)

__device__ __forceinline__ __bf16 f2bf(float f) {
  unsigned u = __builtin_bit_cast(unsigned, f);
  u += 0x7FFFu + ((u >> 16) & 1u);          // round-to-nearest-even
  unsigned short h = (unsigned short)(u >> 16);
  return __builtin_bit_cast(__bf16, h);
}

#define WMMA_BF16(a, b, c) \
  __builtin_amdgcn_wmma_f32_16x16x32_bf16(false, (a), false, (b), (short)0, (c), false, false)

__device__ __forceinline__ v16bf cat8(v8bf lo, v8bf hi) {
  return __builtin_shufflevector(lo, hi, 0, 1, 2, 3, 4, 5, 6, 7,
                                 8, 9, 10, 11, 12, 13, 14, 15);
}

// ---------------------------------------------------------------------------
// One-time f32 -> bf16 conversion (vectorized, grid-stride).
// ---------------------------------------------------------------------------
__global__ __launch_bounds__(256)
void f32_to_bf16(const float* __restrict__ src, __bf16* __restrict__ dst, int n4) {
  int i = blockIdx.x * blockDim.x + threadIdx.x;
  const float4* s4 = (const float4*)src;
  v4bf* d4 = (v4bf*)dst;
  for (; i < n4; i += gridDim.x * blockDim.x) {
    float4 v = s4[i];
    v4bf o;
    o[0] = f2bf(v.x); o[1] = f2bf(v.y); o[2] = f2bf(v.z); o[3] = f2bf(v.w);
    d4[i] = o;
  }
}

// ---------------------------------------------------------------------------
// V transpose: qkv v-slice [B*S, (h,d)] -> vT[b,h,d,key]  (bf16, LDS tiled)
// grid = (S/64, H, B); block = 256. Tile: 64 keys x 64 dims.
// ---------------------------------------------------------------------------
__global__ __launch_bounds__(256)
void transpose_v(const __bf16* __restrict__ qkv, __bf16* __restrict__ vT) {
  __shared__ __bf16 lt[64][72];  // padded rows to spread LDS banks
  const int b = blockIdx.z, h = blockIdx.y;
  const int kt0 = blockIdx.x * 64;
  const __bf16* vbase = qkv + (size_t)b * Sseq * LDQ + h * 192 + 128;

  // read 64x64 tile, key-major (coalesced b128 loads)
  for (int u = threadIdx.x; u < 512; u += 256) {
    const int key = u >> 3, c = u & 7;
    v8bf v = *(const v8bf*)(vbase + (size_t)(kt0 + key) * LDQ + c * 8);
#pragma unroll
    for (int j = 0; j < 8; ++j) lt[key][c * 8 + j] = v[j];
  }
  __syncthreads();
  // write transposed, d-major (coalesced b128 stores)
  __bf16* obase = vT + ((size_t)(b * NH + h) * HDm) * Sseq + kt0;
  for (int u = threadIdx.x; u < 512; u += 256) {
    const int d = u >> 3, kc = u & 7;
    v8bf o;
#pragma unroll
    for (int j = 0; j < 8; ++j) o[j] = lt[kc * 8 + j][d];
    *(v8bf*)(obase + (size_t)d * Sseq + kc * 8) = o;
  }
}

// ---------------------------------------------------------------------------
// GEMM: C[M,N] = A[M,K] * B[N,K]^T    (bf16 in, WMMA f32 accumulate)
// block = 256 threads (8 waves); wave -> 32x64 C strip; block -> 256x64.
// grid = (N/64, M/256). OUT_BF16 selects bf16 or f32 C.
// ---------------------------------------------------------------------------
template <bool OUT_BF16>
__global__ __launch_bounds__(256)
void wmma_gemm_abT(const __bf16* __restrict__ A, const __bf16* __restrict__ B,
                   void* __restrict__ Cv, int M, int N, int K) {
  const int lane = threadIdx.x & 31;
  const int wave = threadIdx.x >> 5;
  const int half = lane >> 4;
  const int l16  = lane & 15;
  const int m0 = blockIdx.y * 256 + wave * 32;
  const int n0 = blockIdx.x * 64;

  v8f acc[2][4] = {};

  const __bf16* ar0 = A + (size_t)(m0 + l16) * K;        // rows m0..m0+15
  const __bf16* ar1 = A + (size_t)(m0 + 16 + l16) * K;   // rows m0+16..m0+31

  for (int k0 = 0; k0 < K; k0 += 32) {
    __builtin_prefetch(ar0 + k0 + 128, 0, 1);
    __builtin_prefetch(ar1 + k0 + 128, 0, 1);
    // A operand (16x32 bf16): lane row = l16; elems K = half*16 + e
    // -> two 16B loads each.
    v16bf a[2];
    a[0] = cat8(*(const v8bf*)(ar0 + k0 + half * 16),
                *(const v8bf*)(ar0 + k0 + half * 16 + 8));
    a[1] = cat8(*(const v8bf*)(ar1 + k0 + half * 16),
                *(const v8bf*)(ar1 + k0 + half * 16 + 8));
#pragma unroll
    for (int nt = 0; nt < 4; ++nt) {
      // B operand (32x16 bf16): lane col = l16; elems K = half*16 + e.
      const __bf16* br = B + (size_t)(n0 + nt * 16 + l16) * K + k0 + half * 16;
      v16bf b = cat8(*(const v8bf*)br, *(const v8bf*)(br + 8));
      acc[0][nt] = WMMA_BF16(a[0], b, acc[0][nt]);
      acc[1][nt] = WMMA_BF16(a[1], b, acc[1][nt]);
    }
  }
  // D layout: VGPR r -> row m0 + mt*16 + half*8 + r, col n0 + nt*16 + l16
#pragma unroll
  for (int mt = 0; mt < 2; ++mt)
#pragma unroll
    for (int nt = 0; nt < 4; ++nt)
#pragma unroll
      for (int r = 0; r < 8; ++r) {
        const size_t idx =
            (size_t)(m0 + mt * 16 + half * 8 + r) * N + n0 + nt * 16 + l16;
        if constexpr (OUT_BF16) ((__bf16*)Cv)[idx] = f2bf(acc[mt][nt][r]);
        else                    ((float*)Cv)[idx]  = acc[mt][nt][r];
      }
}

// ---------------------------------------------------------------------------
// Fused attention for one (b, h, 128-query block).
// grid = (S/128, H, B); block = 256 (8 waves), wave -> 16 query rows.
// qkv: [B*S, 3072] bf16 ; vT: [B,H,64,S] bf16
// attnw: [B,H,S,S] f32 out ; aout: [B*S, 1024] bf16 scratch
// ---------------------------------------------------------------------------
__global__ __launch_bounds__(256)
void attn_fused(const __bf16* __restrict__ qkv, const __bf16* __restrict__ vT,
                float* __restrict__ attnw, __bf16* __restrict__ aout) {
  const int lane = threadIdx.x & 31;
  const int wave = threadIdx.x >> 5;
  const int half = lane >> 4;
  const int l16  = lane & 15;
  const int b = blockIdx.z, h = blockIdx.y;
  const int q0 = blockIdx.x * 128 + wave * 16;
  const float scale = 0.125f;  // 1/sqrt(64)

  const __bf16* qbase = qkv + (size_t)b * Sseq * LDQ + h * 192;
  const __bf16* kbase = qbase + 64;
  // vT row for this lane's head-dim column, per dt tile
  const __bf16* vtb = vT + ((size_t)(b * NH + h) * HDm) * Sseq;

  __shared__ __bf16 pst[8][16 * 32];   // per-wave P staging
  __shared__ float  invs[8][16];

  // ---- Q tile (16 rows x 64 dims) as two WMMA-A chunks --------------------
  v16bf aq[2];
  {
    const __bf16* qrow = qbase + (size_t)(q0 + l16) * LDQ;
#pragma unroll
    for (int c = 0; c < 2; ++c)
      aq[c] = cat8(*(const v8bf*)(qrow + c * 32 + half * 16),
                   *(const v8bf*)(qrow + c * 32 + half * 16 + 8));
  }

  // ---- pass 1: row max of raw scores --------------------------------------
  float mrow[8];
#pragma unroll
  for (int r = 0; r < 8; ++r) mrow[r] = -3.0e38f;

  for (int kt = 0; kt < Sseq; kt += 16) {
    const __bf16* krow = kbase + (size_t)(kt + l16) * LDQ;  // key = kt + l16
    __builtin_prefetch(krow + 16 * LDQ, 0, 1);
    v16bf bk0 = cat8(*(const v8bf*)(krow + half * 16),
                     *(const v8bf*)(krow + half * 16 + 8));
    v16bf bk1 = cat8(*(const v8bf*)(krow + 32 + half * 16),
                     *(const v8bf*)(krow + 32 + half * 16 + 8));
    v8f s = {};
    s = WMMA_BF16(aq[0], bk0, s);
    s = WMMA_BF16(aq[1], bk1, s);
#pragma unroll
    for (int r = 0; r < 8; ++r) mrow[r] = fmaxf(mrow[r], s[r]);
  }
#pragma unroll
  for (int off = 1; off < 16; off <<= 1)
#pragma unroll
    for (int r = 0; r < 8; ++r) mrow[r] = fmaxf(mrow[r], __shfl_xor(mrow[r], off, 32));

  // ---- pass 2: exp, row sums, unnormalized P out, P@V accumulate ----------
  float lrow[8] = {0.f, 0.f, 0.f, 0.f, 0.f, 0.f, 0.f, 0.f};
  v8f pv[4] = {};
  float* attnrow = attnw + ((size_t)(b * NH + h) * Sseq + q0) * Sseq;

  for (int kt = 0; kt < Sseq; kt += 32) {
    v8f s01[2];
#pragma unroll
    for (int t = 0; t < 2; ++t) {
      const __bf16* krow = kbase + (size_t)(kt + t * 16 + l16) * LDQ;
      __builtin_prefetch(krow + 32 * LDQ, 0, 1);
      v16bf bk0 = cat8(*(const v8bf*)(krow + half * 16),
                       *(const v8bf*)(krow + half * 16 + 8));
      v16bf bk1 = cat8(*(const v8bf*)(krow + 32 + half * 16),
                       *(const v8bf*)(krow + 32 + half * 16 + 8));
      v8f s = {};
      s = WMMA_BF16(aq[0], bk0, s);
      s = WMMA_BF16(aq[1], bk1, s);
      s01[t] = s;
    }

    __syncthreads();  // WAR: previous iteration's LDS reads complete
#pragma unroll
    for (int t = 0; t < 2; ++t) {
#pragma unroll
      for (int r = 0; r < 8; ++r) {
        float p = __expf((s01[t][r] - mrow[r]) * scale);
        lrow[r] += p;
        const int row = half * 8 + r;
        attnrow[(size_t)row * Sseq + kt + t * 16 + l16] = p;   // unnormalized
        pst[wave][row * 32 + t * 16 + l16] = f2bf(p);
      }
    }
    __syncthreads();  // RAW: staged P visible before A-layout reload

    // reload P (16x32) in WMMA-A layout from LDS (two ds b128 loads)
    const __bf16* pw = &pst[wave][0];
    v16bf ap = cat8(*(const v8bf*)(pw + l16 * 32 + half * 16),
                    *(const v8bf*)(pw + l16 * 32 + half * 16 + 8));

    // V chunk (32 keys x 64 dims) as four 32x16 B operands from vT:
    // lane col d = dt*16+l16 -> contiguous keys -> two b128 loads per tile.
#pragma unroll
    for (int dt = 0; dt < 4; ++dt) {
      const __bf16* vr = vtb + (size_t)(dt * 16 + l16) * Sseq + kt + half * 16;
      v16bf bv = cat8(*(const v8bf*)vr, *(const v8bf*)(vr + 8));
      pv[dt] = WMMA_BF16(ap, bv, pv[dt]);
    }
  }

  // ---- reduce row sums across the 16-lane half ----------------------------
#pragma unroll
  for (int off = 1; off < 16; off <<= 1)
#pragma unroll
    for (int r = 0; r < 8; ++r) lrow[r] += __shfl_xor(lrow[r], off, 32);
  float inv[8];
#pragma unroll
  for (int r = 0; r < 8; ++r) inv[r] = 1.0f / lrow[r];

  // attn_out tile (scaled, bf16 for the next GEMM)
#pragma unroll
  for (int dt = 0; dt < 4; ++dt)
#pragma unroll
    for (int r = 0; r < 8; ++r)
      aout[(size_t)(b * Sseq + q0 + half * 8 + r) * Dhid + h * HDm + dt * 16 + l16] =
          f2bf(pv[dt][r] * inv[r]);

  // broadcast 1/l to all lanes of the wave
  if (l16 < 8) invs[wave][half * 8 + l16] = inv[l16];
  __syncthreads();

  // ---- pass 3: normalize the (L2-hot) attention-weight strip in place -----
  for (int r = 0; r < 16; ++r) {
    const float iv = invs[wave][r];
    float4* prow = (float4*)(attnrow + (size_t)r * Sseq);
    for (int c = lane; c < Sseq / 4; c += 32) {
      float4 v = prow[c];
      v.x *= iv; v.y *= iv; v.z *= iv; v.w *= iv;
      prow[c] = v;
    }
  }
}

// ---------------------------------------------------------------------------
extern "C" void kernel_launch(void* const* d_in, const int* in_sizes, int n_in,
                              void* d_out, int out_size, void* d_ws, size_t ws_size,
                              hipStream_t stream) {
  const float* x    = (const float*)d_in[0];  // [B,S,Din]
  const float* Wqkv = (const float*)d_in[1];  // [3*Dhid, Din]
  const float* Wo   = (const float*)d_in[2];  // [Dhid, Dhid]

  float* out   = (float*)d_out;                           // [B,S,Dhid]
  float* attnw = out + (size_t)Bsz * Sseq * Dhid;         // [B,H,S,S]

  const int M = Bsz * Sseq;  // 8192

  // workspace layout (bf16)
  __bf16* qkv   = (__bf16*)d_ws;                          // [M, 3072]   48 MB
  __bf16* aoutb = qkv + (size_t)M * LDQ;                  // [M, 1024]   16 MB
  __bf16* xb    = aoutb + (size_t)M * Dhid;               // [M, 1024]   16 MB
  __bf16* wqb   = xb + (size_t)M * Din;                   // [3072,1024]  6 MB
  __bf16* wob   = wqb + (size_t)3 * Dhid * Din;           // [1024,1024]  2 MB
  __bf16* vT    = wob + (size_t)Dhid * Dhid;              // [B,H,64,S]  16 MB

  // 0) one-time f32 -> bf16 conversions
  f32_to_bf16<<<2048, 256, 0, stream>>>(x,    xb,  (M * Din) / 4);
  f32_to_bf16<<<1024, 256, 0, stream>>>(Wqkv, wqb, (3 * Dhid * Din) / 4);
  f32_to_bf16<<<512,  256, 0, stream>>>(Wo,   wob, (Dhid * Dhid) / 4);

  // 1) fused QKV projection: qkv = x @ Wqkv^T   (bf16 out for attention)
  wmma_gemm_abT<true><<<dim3((3 * Dhid) / 64, M / 256), 256, 0, stream>>>(
      xb, wqb, qkv, M, 3 * Dhid, Din);

  // 1b) transpose V so PV operands are unit-stride
  transpose_v<<<dim3(Sseq / 64, NH, Bsz), 256, 0, stream>>>(qkv, vT);

  // 2) fused attention (softmax in-register; weights written once + rescale)
  attn_fused<<<dim3(Sseq / 128, NH, Bsz), 256, 0, stream>>>(qkv, vT, attnw, aoutb);

  // 3) output projection: out = attn_out @ Wo^T  (f32 out)
  wmma_gemm_abT<false><<<dim3(Dhid / 64, M / 256), 256, 0, stream>>>(
      aoutb, wob, out, M, Dhid, Dhid);
}
